// EfficientMultiHeadAttention_60988535603807
// MI455X (gfx1250) — matrix operand order repaired and checked by reading, not compiled, not run on hardware
//
#include <hip/hip_runtime.h>
#include <hip/hip_bf16.h>

// Problem constants
#define B_SZ   4
#define L_SZ   8192
#define D_SZ   1024
#define NHEADS 16
#define HDIM   64
#define MTOT   (B_SZ * L_SZ)      // 32768 rows

// GEMM tiling: 128x128 C-tile, K-step 32, 256 threads = 8 waves (wave32)
#define BM 128
#define BN 128
#define BK 32
#define KT (D_SZ / BK)
#define LDA (BK + 8)              // bf16 elems per As row (80B stride, 16B aligned)
#define LDB (BN + 8)              // bf16 elems per Bs row (272B stride, 16B aligned)

typedef __attribute__((ext_vector_type(16))) __bf16 v16bf;
typedef __attribute__((ext_vector_type(8)))  __bf16 v8bf;
typedef __attribute__((ext_vector_type(8)))  float  v8f;

// ---- CDNA5 async global->LDS (ASYNCcnt) with safe fallback --------------
#if defined(__has_builtin)
#  if __has_builtin(__builtin_amdgcn_global_load_async_to_lds_b128)
#    define HAVE_ASYNC_LDS 1
#  endif
#endif

#if defined(__has_builtin) && __has_builtin(__builtin_amdgcn_s_wait_asynccnt)
#  define WAIT_ASYNC(n) __builtin_amdgcn_s_wait_asynccnt(n)
#else
#  define WAIT_ASYNC(n) asm volatile("s_wait_asynccnt %0" :: "n"(n) : "memory")
#endif

// Builtin signature (from compiler diagnostic): params are int4* in AS1/AS3.
typedef int i32x4 __attribute__((vector_size(16)));
typedef __attribute__((address_space(1))) i32x4* g4p_t;   // global int4*
typedef __attribute__((address_space(3))) i32x4* l4p_t;   // LDS int4*

static __device__ __forceinline__ void copy16_g2lds(const __bf16* g, __bf16* l) {
#ifdef HAVE_ASYNC_LDS
  // generic->AS1 is an identity mapping; generic LDS ptr low 32 bits == AS3 offset
  g4p_t gp = (g4p_t)(unsigned long long)g;
  l4p_t lp = (l4p_t)(unsigned int)(unsigned long long)l;
  __builtin_amdgcn_global_load_async_to_lds_b128(gp, lp, 0, 0);
#else
  *(v8bf*)l = *(const v8bf*)g;
#endif
}

static __device__ __forceinline__ __bf16 f2bf(float f) {
  unsigned u = __builtin_bit_cast(unsigned, f);
  unsigned r = (u + 0x7FFFu + ((u >> 16) & 1u)) >> 16;   // round-to-nearest-even
  unsigned short s = (unsigned short)r;
  return __builtin_bit_cast(__bf16, s);
}

static __device__ __forceinline__ v16bf cat8(v8bf lo, v8bf hi) {
  return __builtin_shufflevector(lo, hi, 0,1,2,3,4,5,6,7,8,9,10,11,12,13,14,15);
}

// phi(x) = elu(x) + 1  =  x+1 (x>0) else exp(x)
static __device__ __forceinline__ float phi_act(float x) {
  return x > 0.f ? x + 1.f : __expf(x);
}

// ---------------------------------------------------------------------------
// Kernel 0: one-shot f32 -> bf16 conversion (8 elements / thread)
// ---------------------------------------------------------------------------
__global__ __launch_bounds__(256)
void emha_cvt_bf16(const float* __restrict__ src, __bf16* __restrict__ dst) {
  const size_t i8 = (size_t)blockIdx.x * 256 + threadIdx.x;
  const float4 a = ((const float4*)src)[i8 * 2];
  const float4 b = ((const float4*)src)[i8 * 2 + 1];
  v8bf h;
  h[0] = f2bf(a.x); h[1] = f2bf(a.y); h[2] = f2bf(a.z); h[3] = f2bf(a.w);
  h[4] = f2bf(b.x); h[5] = f2bf(b.y); h[6] = f2bf(b.z); h[7] = f2bf(b.w);
  *(v8bf*)(dst + i8 * 8) = h;
}

// ---------------------------------------------------------------------------
// Kernel 1: fused QKV GEMM (z selects Q/K/V), bf16 WMMA, fp32 accumulate,
// async global->LDS double-buffered pipeline.
//   z=0: phiQ = elu(X@Wq+bq)+1            -> d_out
//   z=1: phiK = mask ? 0 : elu(X@Wk+bk)+1 -> ws
//   z=2: Vsum[row,head] = sum_v (X@Wv+bv) masked -> ws  (V never stored)
// ---------------------------------------------------------------------------
__global__ __launch_bounds__(256, 2)
void emha_qkv_gemm(const __bf16* __restrict__ Xbf,
                   const __bf16* __restrict__ Wbf,   // 3 contiguous DxD matrices
                   const float* __restrict__ bq, const float* __restrict__ bk,
                   const float* __restrict__ bv,
                   const int*   __restrict__ mask,
                   float* __restrict__ phiQ,
                   float* __restrict__ phiK,
                   float* __restrict__ Vsum)
{
  const int zid = blockIdx.z;
  const __bf16* __restrict__ W   = Wbf + (size_t)zid * D_SZ * D_SZ;
  const float* __restrict__ bias = (zid == 0) ? bq : (zid == 1 ? bk : bv);

  const int n0 = blockIdx.x * BN;
  const int m0 = blockIdx.y * BM;

  __shared__ __bf16 As[2][BM][LDA];
  __shared__ __bf16 Bs[2][BK][LDB];
  __shared__ float  red[BM][2][2];   // V epilogue: [row][head-in-block][wave-pair]

  const int t    = threadIdx.x;
  const int lane = t & 31;
  const int w    = t >> 5;
  const int wm   = w >> 2;           // 0..1 -> rows wm*64 .. +63
  const int wn   = w & 3;            // 0..3 -> cols wn*32 .. +31

  // Per-thread staging coordinates (4 x 16B copies per tile)
  const int arow0 = t >> 2,         aseg0 = (t & 3) * 8;         // A chunk 0
  const int arow1 = (t + 256) >> 2, aseg1 = aseg0;               // A chunk 1
  const int brow0 = t >> 4,         bseg0 = (t & 15) * 8;        // B chunk 0
  const int brow1 = (t + 256) >> 4, bseg1 = bseg0;               // B chunk 1

  auto stage = [&](int kt, int buf) {
    copy16_g2lds(Xbf + (size_t)(m0 + arow0) * D_SZ + kt * BK + aseg0, &As[buf][arow0][aseg0]);
    copy16_g2lds(Xbf + (size_t)(m0 + arow1) * D_SZ + kt * BK + aseg1, &As[buf][arow1][aseg1]);
    copy16_g2lds(W + (size_t)(kt * BK + brow0) * D_SZ + n0 + bseg0, &Bs[buf][brow0][bseg0]);
    copy16_g2lds(W + (size_t)(kt * BK + brow1) * D_SZ + n0 + bseg1, &Bs[buf][brow1][bseg1]);
  };

  v8f acc[4][2];
  #pragma unroll
  for (int i = 0; i < 4; ++i)
    #pragma unroll
    for (int j = 0; j < 2; ++j)
      acc[i][j] = (v8f){0.f,0.f,0.f,0.f,0.f,0.f,0.f,0.f};

  stage(0, 0);
  for (int kt = 0; kt < KT; ++kt) {
    const int buf = kt & 1;
    if (kt + 1 < KT) {                 // prefetch next tile into other buffer
      stage(kt + 1, buf ^ 1);
      WAIT_ASYNC(4);                   // my 4 current-tile copies have landed
    } else {
      WAIT_ASYNC(0);
    }
    __syncthreads();                   // all waves' copies have landed

    // ---- A fragments: lane<16 -> K{0..7,16..23}; lane>=16 -> K{8..15,24..31}
    const int lrow  = lane & 15;
    const int khalf = (lane < 16) ? 0 : 8;
    v16bf af[4];
    #pragma unroll
    for (int i = 0; i < 4; ++i) {
      const __bf16* p = &As[buf][wm * 64 + i * 16 + lrow][khalf];
      af[i] = cat8(*(const v8bf*)p, *(const v8bf*)(p + 16));
    }
    // ---- B fragments: lane holds column N=lane&15, K{0..7,16..23} / K{8..15,24..31} like A
    v16bf bfm[2];
    #pragma unroll
    for (int j = 0; j < 2; ++j) {
      const __bf16* p = &Bs[buf][khalf][wn * 32 + j * 16 + lrow];
      #pragma unroll
      for (int s = 0; s < 16; ++s)
        bfm[j][s] = p[(s + (s & 8)) * LDB];
    }
    // ---- 8 WMMAs per K-step
    #pragma unroll
    for (int i = 0; i < 4; ++i)
      #pragma unroll
      for (int j = 0; j < 2; ++j)
        acc[i][j] = __builtin_amdgcn_wmma_f32_16x16x32_bf16(
            false, af[i], false, bfm[j], (short)0, acc[i][j], false, false);
    __syncthreads();                   // done reading buf before it is re-filled
  }

  // ---- epilogue: C layout = VGPR r: lanes0-15 M=r, lanes16-31 M=8+r; N=lane&15
  const int rhi = (lane >> 4) << 3;
  if (zid < 2) {
    float* __restrict__ dst = (zid == 0) ? phiQ : phiK;
    #pragma unroll
    for (int i = 0; i < 4; ++i) {
      #pragma unroll
      for (int j = 0; j < 2; ++j) {
        const int col  = n0 + wn * 32 + j * 16 + (lane & 15);
        const float bc = bias[col];
        const int rb   = m0 + wm * 64 + i * 16 + rhi;
        #pragma unroll
        for (int r = 0; r < 8; ++r) {
          const int row = rb + r;
          float x = acc[i][j][r] + bc;
          float y = phi_act(x);
          if (zid == 1 && mask[row]) y = 0.f;   // elu(-1e10)+1 == 0
          dst[(size_t)row * D_SZ + col] = y;
        }
      }
    }
  } else {
    // V: per-head row sums. Wave covers 32 cols = half a head.
    const int head = wn >> 1, part = wn & 1;
    const float b0 = bias[n0 + wn * 32 + (lane & 15)];
    const float b1 = bias[n0 + wn * 32 + 16 + (lane & 15)];
    #pragma unroll
    for (int i = 0; i < 4; ++i) {
      #pragma unroll
      for (int r = 0; r < 8; ++r) {
        float s = (acc[i][0][r] + b0) + (acc[i][1][r] + b1);
        s += __shfl_xor(s, 1, 32);   // reduce across 16 cols; halves stay separate
        s += __shfl_xor(s, 2, 32);
        s += __shfl_xor(s, 4, 32);
        s += __shfl_xor(s, 8, 32);
        if ((lane & 15) == 0)
          red[wm * 64 + i * 16 + rhi + r][head][part] = s;
      }
    }
    __syncthreads();
    {
      const int rowloc = t >> 1, h = t & 1;
      const int row = m0 + rowloc;
      float vs = red[rowloc][h][0] + red[rowloc][h][1];
      if (mask[row]) vs = 0.f;
      Vsum[(size_t)row * NHEADS + (n0 / HDIM) + h] = vs;
    }
  }
}

// ---------------------------------------------------------------------------
// Kernel 2: per-chunk partial kv_sum[b,n,k] = sum_l phiK[l,n*64+k] * Vsum[l,n]
// ---------------------------------------------------------------------------
__global__ __launch_bounds__(256)
void emha_kv_partial(const float* __restrict__ phiK,
                     const float* __restrict__ Vsum,
                     float* __restrict__ partial)
{
  const int chunk = blockIdx.x;        // 0..63
  const int b     = blockIdx.y;        // 0..3
  const int t     = threadIdx.x;
  float a0 = 0.f, a1 = 0.f, a2 = 0.f, a3 = 0.f;
  const int base_row = b * L_SZ + chunk * 128;
  for (int rr = 0; rr < 128; ++rr) {
    const int row = base_row + rr;
    const float* __restrict__ pk = phiK + (size_t)row * D_SZ;
    const float* __restrict__ vs = Vsum + (size_t)row * NHEADS;
    __builtin_prefetch(pk + 4 * D_SZ + t, 0, 1);     // stream ahead
    a0 += pk[t      ] * vs[(t      ) >> 6];
    a1 += pk[t + 256] * vs[(t + 256) >> 6];
    a2 += pk[t + 512] * vs[(t + 512) >> 6];
    a3 += pk[t + 768] * vs[(t + 768) >> 6];
  }
  float* dst = partial + (size_t)(b * 64 + chunk) * D_SZ;
  dst[t] = a0; dst[t + 256] = a1; dst[t + 512] = a2; dst[t + 768] = a3;
}

// Kernel 3: deterministic reduction over 64 chunks -> kv_sum[b][1024]
__global__ __launch_bounds__(256)
void emha_kv_reduce(const float* __restrict__ partial, float* __restrict__ kvsum)
{
  const int idx = blockIdx.x * 256 + threadIdx.x;   // 0..4095
  const int b = idx >> 10, c = idx & 1023;
  float s = 0.f;
  for (int ch = 0; ch < 64; ++ch)
    s += partial[(size_t)(b * 64 + ch) * D_SZ + c];
  kvsum[idx] = s;
}

// Kernel 4: out[row][c] = phiQ[row][c] * kv_sum[b][c]  (in place on d_out)
__global__ __launch_bounds__(256)
void emha_scale_out(float* __restrict__ out, const float* __restrict__ kvsum)
{
  const int i4  = blockIdx.x * 256 + threadIdx.x;   // float4 index
  const int row = i4 >> 8;                          // 256 float4 per row
  const int c4  = i4 & 255;
  const int b   = row >> 13;                        // L = 8192
  float4 o = ((float4*)out)[i4];
  const float4 kv = ((const float4*)(kvsum + (size_t)b * D_SZ))[c4];
  o.x *= kv.x; o.y *= kv.y; o.z *= kv.z; o.w *= kv.w;
  ((float4*)out)[i4] = o;
}

extern "C" void kernel_launch(void* const* d_in, const int* in_sizes, int n_in,
                              void* d_out, int out_size, void* d_ws, size_t ws_size,
                              hipStream_t stream) {
  const float* X    = (const float*)d_in[0];
  const int*   mask = (const int*)  d_in[1];
  const float* Wq   = (const float*)d_in[2];
  const float* bq   = (const float*)d_in[3];
  const float* Wk   = (const float*)d_in[4];
  const float* bk   = (const float*)d_in[5];
  const float* Wv   = (const float*)d_in[6];
  const float* bv   = (const float*)d_in[7];
  float* out = (float*)d_out;

  // Workspace layout:
  //   phiK   : MTOT*D   f32  (128 MB)
  //   Vsum   : MTOT*16  f32  (2 MB)
  //   partial: 256*1024 f32  (1 MB)
  //   kvsum  : 4096     f32  (16 KB)
  //   Xbf    : MTOT*D   bf16 (64 MB)
  //   Wbf    : 3*D*D    bf16 (6 MB)
  float* ws      = (float*)d_ws;
  float* phiK    = ws;
  float* Vsum    = phiK + (size_t)MTOT * D_SZ;
  float* partial = Vsum + (size_t)MTOT * NHEADS;
  float* kvsum   = partial + (size_t)256 * D_SZ;
  __bf16* Xbf    = (__bf16*)(kvsum + 4096);
  __bf16* Wbf    = Xbf + (size_t)MTOT * D_SZ;

  // 0) one-shot f32 -> bf16 conversions
  emha_cvt_bf16<<<(MTOT * D_SZ) / 2048, 256, 0, stream>>>(X, Xbf);
  emha_cvt_bf16<<<(D_SZ * D_SZ) / 2048, 256, 0, stream>>>(Wq, Wbf);
  emha_cvt_bf16<<<(D_SZ * D_SZ) / 2048, 256, 0, stream>>>(Wk, Wbf + (size_t)D_SZ * D_SZ);
  emha_cvt_bf16<<<(D_SZ * D_SZ) / 2048, 256, 0, stream>>>(Wv, Wbf + (size_t)2 * D_SZ * D_SZ);

  // 1) fused QKV WMMA GEMM
  dim3 gA(D_SZ / BN, MTOT / BM, 3);   // (8, 256, 3)
  emha_qkv_gemm<<<gA, 256, 0, stream>>>(Xbf, Wbf, bq, bk, bv, mask,
                                        out, phiK, Vsum);
  // 2-4) linear-attention reduction + output scale
  emha_kv_partial<<<dim3(64, 4), 256, 0, stream>>>(phiK, Vsum, partial);
  emha_kv_reduce<<<16, 256, 0, stream>>>(partial, kvsum);
  emha_scale_out<<<(MTOT * (D_SZ / 4)) / 256, 256, 0, stream>>>(out, kvsum);
}
